// SoftmaxConvertBias_65936337928357
// MI455X (gfx1250) — compile-verified, matrix operand order checked by reading, not run on hardware
//
#include <hip/hip_runtime.h>
#include <math.h>

// ---------------- problem constants ----------------
#define VOCAB      32000
#define VP1        32001          // V + 1 (mask slot)
#define VPAD       32128          // V+1 padded to multiple of 128
#define DNOISE     1024
#define DBB        4096
#define ROWS       4096           // B*T

// ---------------- types ----------------
typedef unsigned int  u32x4  __attribute__((ext_vector_type(4)));
typedef __bf16        bf16x16 __attribute__((ext_vector_type(16)));
typedef float         f32x8  __attribute__((ext_vector_type(8)));

union Frag { bf16x16 v; u32x4 u[2]; };

// async global->LDS path (gfx1250 GLOBAL_LOAD_ASYNC_TO_LDS_B128, ASYNCcnt-tracked)
#if __has_builtin(__builtin_amdgcn_global_load_async_to_lds_b128) && \
    __has_builtin(__builtin_amdgcn_s_wait_asynccnt)
#define HAS_ASYNC_LDS 1
// builtin prototype (from hipcc diagnostic): pointee is GCC int 4-vector,
// src in addrspace(1) ("__device__"), dst in addrspace(3)
typedef int i32x4v __attribute__((vector_size(4 * sizeof(int))));
typedef __attribute__((address_space(1))) i32x4v as1_i32x4;
typedef __attribute__((address_space(3))) i32x4v as3_i32x4;
#else
#define HAS_ASYNC_LDS 0
#endif

__device__ __forceinline__ unsigned short f32_to_bf16(float f) {
    unsigned int u = __float_as_uint(f);
    u = (u + 0x7FFFu + ((u >> 16) & 1u)) >> 16;   // round-to-nearest-even
    return (unsigned short)u;
}

// ---------------- conversion kernels ----------------
__global__ __launch_bounds__(256) void cvt_bf16_kernel(const float* __restrict__ in,
                                                       unsigned short* __restrict__ out, int n) {
    int i = blockIdx.x * 256 + threadIdx.x;
    if (i < n) out[i] = f32_to_bf16(in[i]);
}

// K table: rows 0..31999 = embed_w, rows 32000..32127 = 0 (mask slot + pad)
__global__ __launch_bounds__(256) void build_k_kernel(const float* __restrict__ embed,
                                                      unsigned short* __restrict__ Kb) {
    int col = blockIdx.x * 256 + threadIdx.x;   // 0..1023 (grid.x = 4)
    int row = blockIdx.y;                        // 0..VPAD-1
    float v = (row < VOCAB) ? embed[(size_t)row * DNOISE + col] : 0.0f;
    Kb[(size_t)row * DNOISE + col] = f32_to_bf16(v);
}

// W_bb: (4096 x 32001) row-major -> bf16 (4096 x VPAD), pad cols zeroed
__global__ __launch_bounds__(256) void build_w_kernel(const float* __restrict__ W,
                                                      unsigned short* __restrict__ Wb) {
    int col = blockIdx.x * 256 + threadIdx.x;   // grid.x = 126 -> guard
    int row = blockIdx.y;                        // 0..4095
    if (col < VPAD) {
        float v = (col < VP1) ? W[(size_t)row * VP1 + col] : 0.0f;
        Wb[(size_t)row * VPAD + col] = f32_to_bf16(v);
    }
}

__global__ __launch_bounds__(256) void build_bias_kernel(const float* __restrict__ b,
                                                         float* __restrict__ bp) {
    for (int i = threadIdx.x; i < VPAD; i += 256)
        bp[i] = (i < VP1) ? b[i] : -1.0e30f;
}

// ---------------- WMMA GEMM: C[M x N] = A[M x K] * Bt[N x K]^T ----------------
// Block tile 128x128, BK=32, 8 waves (2 M x 4 N), each wave 4x2 subtiles of 16x16.
// EPI==1: C = beta*acc + bias[col]   (logits epilogue)
// EPI==0: C = acc + bias[col]        (backbone epilogue)
#define LDT 48   // padded LDS row stride (elements); 96B rows keep b128 aligned

template<int EPI>
__global__ __launch_bounds__(256) void gemm_wmma_kernel(
        const unsigned short* __restrict__ A,   // M x K bf16 row-major
        const unsigned short* __restrict__ Bt,  // N x K bf16 row-major
        float* __restrict__ C, int K, int ldc,
        const float* __restrict__ scale,        // beta (EPI==1)
        const float* __restrict__ bias)         // per-column bias
{
    __shared__ unsigned short As[128 * LDT];
    __shared__ unsigned short Bs[128 * LDT];

    const int tid   = threadIdx.x;
    const int lane  = tid & 31;
    const int wave  = tid >> 5;
    const int waveM = wave & 1;        // 2 waves over M
    const int waveN = wave >> 1;       // 4 waves over N
    const int lrow  = lane & 15;       // M/N index within 16x16 tile
    const int lhalf = lane >> 4;       // K-half selector (ISA fragment layout)

    const int blockM = blockIdx.x * 128;
    const int blockN = blockIdx.y * 128;

    f32x8 acc[4][2];
    const f32x8 zacc = {0.f,0.f,0.f,0.f,0.f,0.f,0.f,0.f};
#pragma unroll
    for (int i = 0; i < 4; ++i)
#pragma unroll
        for (int j = 0; j < 2; ++j) acc[i][j] = zacc;

    // global->LDS staging: thread t copies 8 elems (16B) of row (t/2), k offset 16*(t&1)
    const int cr = tid >> 1;                 // 0..127
    const int ch = (tid & 1) * 16;           // 0 or 16
    const unsigned short* aSrc = A  + (size_t)(blockM + cr) * K + ch;
    const unsigned short* bSrc = Bt + (size_t)(blockN + cr) * K + ch;
    unsigned short* aDst = &As[cr * LDT + ch];
    unsigned short* bDst = &Bs[cr * LDT + ch];

    for (int kk = 0; kk < K; kk += 32) {
#if HAS_ASYNC_LDS
        __syncthreads();                       // previous iteration's frags consumed
        // per-lane 16B DMA straight into LDS, tracked by ASYNCcnt
        __builtin_amdgcn_global_load_async_to_lds_b128(
            (as1_i32x4*)(aSrc + kk), (as3_i32x4*)aDst, 0, 0);
        __builtin_amdgcn_global_load_async_to_lds_b128(
            (as1_i32x4*)(bSrc + kk), (as3_i32x4*)bDst, 0, 0);
        __builtin_amdgcn_s_wait_asynccnt(0);   // this wave's copies landed
        __syncthreads();                       // all waves' copies landed
#else
        u32x4 av = *(const u32x4*)(aSrc + kk);
        u32x4 bv = *(const u32x4*)(bSrc + kk);
        __syncthreads();                       // previous iteration's frags consumed
        *(u32x4*)aDst = av;
        *(u32x4*)bDst = bv;
        __syncthreads();
#endif

        if (kk + 32 < K) {                     // global_prefetch_b8 for next K-slice
            __builtin_prefetch((const void*)(aSrc + kk + 32), 0, 1);
            __builtin_prefetch((const void*)(bSrc + kk + 32), 0, 1);
        }

        Frag aF[4], bF[2];
#pragma unroll
        for (int i = 0; i < 4; ++i) {
            const unsigned short* p = &As[(waveM * 64 + i * 16 + lrow) * LDT + lhalf * 8];
            aF[i].u[0] = *(const u32x4*)p;          // K = h*8 .. h*8+7
            aF[i].u[1] = *(const u32x4*)(p + 16);   // K = h*8+16 .. h*8+23
        }
#pragma unroll
        for (int j = 0; j < 2; ++j) {
            const unsigned short* p = &Bs[(waveN * 32 + j * 16 + lrow) * LDT + lhalf * 8];
            bF[j].u[0] = *(const u32x4*)p;
            bF[j].u[1] = *(const u32x4*)(p + 16);
        }
#pragma unroll
        for (int i = 0; i < 4; ++i)
#pragma unroll
            for (int j = 0; j < 2; ++j)
                acc[i][j] = __builtin_amdgcn_wmma_f32_16x16x32_bf16(
                    false, aF[i].v, false, bF[j].v, (short)0, acc[i][j], false, false);
    }

    const float betaVal = (EPI == 1) ? scale[0] : 1.0f;
#pragma unroll
    for (int i = 0; i < 4; ++i) {
        const int rowBase = blockM + waveM * 64 + i * 16 + lhalf * 8;
#pragma unroll
        for (int j = 0; j < 2; ++j) {
            const int col = blockN + waveN * 32 + j * 16 + lrow;
            const float bval = bias[col];
#pragma unroll
            for (int r = 0; r < 8; ++r) {
                float v = acc[i][j][r];
                v = (EPI == 1) ? (v * betaVal + bval) : (v + bval);
                C[(size_t)(rowBase + r) * ldc + col] = v;
            }
        }
    }
}

// ---------------- fp32 online softmax, bf16 probs out ----------------
__global__ __launch_bounds__(256) void softmax_kernel(const float* __restrict__ logits,
                                                      unsigned short* __restrict__ probs) {
    const int row = blockIdx.x;
    const float* x = logits + (size_t)row * VPAD;
    unsigned short* p = probs + (size_t)row * VPAD;
    const int tid = threadIdx.x;

    float m = -INFINITY, s = 0.0f;
    for (int i = tid; i < VP1; i += 256) {
        float v  = x[i];
        float mn = fmaxf(m, v);
        s = s * __expf(m - mn) + __expf(v - mn);
        m = mn;
    }
#pragma unroll
    for (int off = 16; off; off >>= 1) {
        float mo = __shfl_xor(m, off, 32);
        float so = __shfl_xor(s, off, 32);
        float mn = fmaxf(m, mo);
        s = s * __expf(m - mn) + so * __expf(mo - mn);
        m = mn;
    }
    __shared__ float wm[8], ws[8];
    const int wv = tid >> 5, ln = tid & 31;
    if (ln == 0) { wm[wv] = m; ws[wv] = s; }
    __syncthreads();
    if (tid == 0) {
        float M = wm[0], S = ws[0];
        for (int w = 1; w < 8; ++w) {
            float mn = fmaxf(M, wm[w]);
            S = S * __expf(M - mn) + ws[w] * __expf(wm[w] - mn);
            M = mn;
        }
        wm[0] = M; ws[0] = 1.0f / S;
    }
    __syncthreads();
    const float M = wm[0], invS = ws[0];
    for (int i = tid; i < VPAD; i += 256) {
        float v = (i < VP1) ? __expf(x[i] - M) * invS : 0.0f;
        p[i] = f32_to_bf16(v);
    }
}

// ---------------- workspace layout (bytes, all 256-aligned) ----------------
#define WS_ZB     0ull
#define WS_KB     (WS_ZB  + (size_t)ROWS * DNOISE * 2)        //   8,388,608
#define WS_WB     (WS_KB  + (size_t)VPAD * DNOISE * 2)        //  74,186,752
#define WS_BIAS   (WS_WB  + (size_t)DBB  * VPAD   * 2)        // 337,379,328
#define WS_LOGITS (WS_BIAS + (size_t)VPAD * 4)                // 337,507,840
#define WS_PROBS  (WS_LOGITS + (size_t)ROWS * VPAD * 4)       // 863,892,992
// total ~1.13 GB

extern "C" void kernel_launch(void* const* d_in, const int* in_sizes, int n_in,
                              void* d_out, int out_size, void* d_ws, size_t ws_size,
                              hipStream_t stream) {
    const float* z          = (const float*)d_in[0];
    const float* embed_w    = (const float*)d_in[1];
    const float* W_bb       = (const float*)d_in[2];
    const float* b_bb       = (const float*)d_in[3];
    const float* beta       = (const float*)d_in[4];
    const float* logit_bias = (const float*)d_in[5];
    float* out = (float*)d_out;

    char* ws = (char*)d_ws;
    unsigned short* zb      = (unsigned short*)(ws + WS_ZB);
    unsigned short* Kb      = (unsigned short*)(ws + WS_KB);
    unsigned short* Wb      = (unsigned short*)(ws + WS_WB);
    float*          biasPad = (float*)         (ws + WS_BIAS);
    float*          logits  = (float*)         (ws + WS_LOGITS);
    unsigned short* probs   = (unsigned short*)(ws + WS_PROBS);

    // operand conversion to bf16 (+ padding / mask-slot zeroing)
    cvt_bf16_kernel<<<(ROWS * DNOISE) / 256, 256, 0, stream>>>(z, zb, ROWS * DNOISE);
    build_k_kernel<<<dim3(DNOISE / 256, VPAD), 256, 0, stream>>>(embed_w, Kb);
    build_w_kernel<<<dim3((VPAD + 255) / 256, DBB), 256, 0, stream>>>(W_bb, Wb);
    build_bias_kernel<<<1, 256, 0, stream>>>(logit_bias, biasPad);

    // GEMM1: logits = beta * (z @ K^T) + bias     (M=4096, N=VPAD, K=1024)
    gemm_wmma_kernel<1><<<dim3(ROWS / 128, VPAD / 128), 256, 0, stream>>>(
        zb, Kb, logits, DNOISE, VPAD, beta, biasPad);

    // row-wise fp32 softmax -> bf16 probs
    softmax_kernel<<<ROWS, 256, 0, stream>>>(logits, probs);

    // GEMM2: out = probs @ W_bb^T + b_bb          (M=4096, N=4096, K=VPAD)
    gemm_wmma_kernel<0><<<dim3(ROWS / 128, DBB / 128), 256, 0, stream>>>(
        probs, Wb, out, VPAD, DBB, nullptr, b_bb);
}
